// naivePC_55963423867088
// MI455X (gfx1250) — matrix operand order repaired and checked by reading, not compiled, not run on hardware
//
#include <hip/hip_runtime.h>
#include <stdint.h>

typedef __attribute__((ext_vector_type(16))) _Float16 v16h;
typedef __attribute__((ext_vector_type(8)))  float    v8f;

#define N_   20
#define NP1  21

// d_ws layout:
//   [0,    1024)  : WMMA B-operand table (32 lanes x 16 f16) = wsm.T cols 0..15
//   [2048, 3560)  : w2 softmax pairs, f32 [9][21][2]

// ---------------------------------------------------------------------------
// Prep: masked softmax of Wleaf -> B-operand lane layout (single N-tile,
// cols 0..15; popcount==8 forces p2<=13 so higher columns are never read);
// pairwise softmax of W2 -> (w0,w1).
// ---------------------------------------------------------------------------
__global__ void prep_kernel(const float* __restrict__ Wleaf,
                            const float* __restrict__ W2,
                            _Float16* __restrict__ btab,
                            float* __restrict__ w2s) {
    __shared__ float wsm[NP1][N_];
    const int t = threadIdx.x;            // 32 threads
    if (t < NP1) {
        const int m = t;
        for (int i = 0; i < N_; ++i) wsm[m][i] = 0.0f;
        if (m > 0) {
            float mx = -1e30f;
            for (int i = 0; i < m; ++i) mx = fmaxf(mx, Wleaf[m * N_ + i]);
            float s = 0.0f;
            for (int i = 0; i < m; ++i) {
                float e = expf(Wleaf[m * N_ + i] - mx);
                wsm[m][i] = e; s += e;
            }
            const float inv = 1.0f / s;
            for (int i = 0; i < m; ++i) wsm[m][i] *= inv;
        }
    }
    __syncthreads();

    // Dense f16 B 32x16 operand layout: element e of lane L holds
    // Bmat[K][n] with K = (e<8 ? 0 : 16) + (L<16?0:8) + (e&7), n = L%16.
    const int  lane = t;
    const bool hi   = lane >= 16;
    const int  n    = lane & 15;
    for (int e = 0; e < 16; ++e) {
        const int K = ((e < 8) ? 0 : 16) + (hi ? 8 : 0) + (e & 7);
        float val = 0.0f;
        if (K < N_ && K < n) val = wsm[n][K];     // softmax mask: i < m
        btab[lane * 16 + e] = (_Float16)val;
    }

    for (int idx = t; idx < 9 * NP1; idx += 32) { // k in 0..8, m in 0..20
        const float a = W2[idx * 2 + 0], b = W2[idx * 2 + 1];
        const float w0 = 1.0f / (1.0f + expf(b - a));
        w2s[idx * 2 + 0] = w0;
        w2s[idx * 2 + 1] = 1.0f - w0;
    }
}

// ---------------------------------------------------------------------------
// Main: WMMA computes dots = xf @ wsm.T for 32 rows/wave (leaf bases);
// per-lane DP collapses to a single staircase path product.
// ---------------------------------------------------------------------------
__global__ void __launch_bounds__(256)
pc_kernel(const unsigned char* __restrict__ x,
          const _Float16* __restrict__ btab,
          const float* __restrict__ w2s,
          float* __restrict__ out, int Bsz) {
    __shared__ float sdots[8 * 32 * 17];          // 8 waves x 32 rows, stride 17 (odd)
    __shared__ float lw2[NP1 * NP1 * 2];          // padded: any k<=20,m<=20 in-bounds

    const int  lane = threadIdx.x & 31;
    const int  wave = threadIdx.x >> 5;
    const int  rb   = blockIdx.x * 256 + wave * 32;
    const bool hi   = lane >= 16;
    const int  lm   = lane & 15;

    // ---- load x bytes for the two M-tile rows this lane feeds into A ----
    int r0 = rb + lm;
    int r1 = r0 + 16;
    if (r0 >= Bsz) r0 = Bsz - 1;
    if (r1 >= Bsz) r1 = Bsz - 1;
    uint32_t d0[5], d1[5];                        // rows are 4B aligned (20%4==0)
    const uint32_t* xr0 = (const uint32_t*)(x + (size_t)r0 * N_);
    const uint32_t* xr1 = (const uint32_t*)(x + (size_t)r1 * N_);
#pragma unroll
    for (int i = 0; i < 5; ++i) { d0[i] = xr0[i]; d1[i] = xr1[i]; }

    // ---- f16 A operands per ISA 16-bit A 16x32 layout ----
    auto bldA = [&](const uint32_t d[5]) -> v16h {
        union { v16h v; unsigned short s[16]; } u;
#pragma unroll
        for (int e = 0; e < 16; ++e) {
            const int Klo = (e < 8) ? e : (16 + (e - 8));
            const int Khi = (e < 8) ? (8 + e) : (24 + (e - 8));
            unsigned short vlo = 0, vhi = 0;
            if (Klo < N_) vlo = ((d[Klo >> 2] >> ((Klo & 3) * 8)) & 0xFFu) ? (unsigned short)0x3C00 : (unsigned short)0;
            if (Khi < N_) vhi = ((d[Khi >> 2] >> ((Khi & 3) * 8)) & 0xFFu) ? (unsigned short)0x3C00 : (unsigned short)0;
            u.s[e] = hi ? vhi : vlo;
        }
        return u.v;
    };
    const v16h a0 = bldA(d0);
    const v16h a1 = bldA(d1);

    v16h b0;
    {
        union { uint4 q[2]; v16h v; } u;
        const uint4* p = (const uint4*)btab;
        u.q[0] = p[lane * 2 + 0]; u.q[1] = p[lane * 2 + 1];
        b0 = u.v;
    }

    // ---- 2x WMMA: dots[32 rows][16 cols] f32 ----
    v8f c0 = {}, c1 = {};
    c0 = __builtin_amdgcn_wmma_f32_16x16x32_f16(false, a0, false, b0, (short)0, c0, false, false);
    c1 = __builtin_amdgcn_wmma_f32_16x16x32_f16(false, a1, false, b0, (short)0, c1, false, false);

    // ---- transpose through LDS ----
    float* wl = &sdots[wave * (32 * 17)];
    const int rbase = hi ? 8 : 0;                 // C layout: VGPR v -> M = v (+8 hi)
#pragma unroll
    for (int v = 0; v < 8; ++v) {
        wl[(rbase + v) * 17 + lm]      = c0[v];
        wl[(16 + rbase + v) * 17 + lm] = c1[v];
    }
    for (int i = threadIdx.x; i < 9 * NP1 * 2; i += 256) lw2[i] = w2s[i];
    __syncthreads();

    // ---- per-lane closed-form path product ----
    uint32_t dw[5];
#pragma unroll
    for (int i = 0; i < 5; ++i) dw[i] = hi ? d1[i] : d0[i];
    uint32_t msk = 0;
#pragma unroll
    for (int j = 0; j < N_; ++j)
        msk |= ((dw[j >> 2] >> ((j & 3) * 8)) & 1u) << j;

    const float* myd = wl + lane * 17;
    const int      cnt   = __popc(msk);
    const uint32_t t2    = msk & (msk - 1u);      // mask minus lowest set bit
    const int      jones = __builtin_ctz(~msk);   // leading-ones run length

    int m0; float base;
    if (jones >= 2) {                             // cumprod base at (j,j), pre == 1
        m0 = jones; base = 1.0f;
    } else {                                      // leaf base dots[p2]
        int p2 = t2 ? (int)__builtin_ctz(t2) : 2; // 2nd set bit; clamps only hit
        if (p2 < 2)  p2 = 2;                      //   gated (cnt!=8) lanes
        if (p2 > 13) p2 = 13;
        m0 = p2; base = myd[p2];
    }

    float cur = base;
    int k = (int)(msk & 1u);                      // popcount(first 1 bit)
#pragma unroll
    for (int m = 2; m <= N_; ++m) {
        const uint32_t bit = (msk >> (m - 1)) & 1u;
        k += bit;                                 // k_m = popcount(first m bits)
        const float w = lw2[(unsigned)(k * NP1 + m) * 2u + bit];
        cur *= (m > m0) ? w : 1.0f;
    }

    const int row = rb + lane;
    if (row < Bsz) out[row] = (cnt == 8) ? cur : 0.0f;
}

// ---------------------------------------------------------------------------
extern "C" void kernel_launch(void* const* d_in, const int* in_sizes, int n_in,
                              void* d_out, int out_size, void* d_ws, size_t ws_size,
                              hipStream_t stream) {
    const unsigned char* x     = (const unsigned char*)d_in[0];   // bool (B,20)
    const float*         Wleaf = (const float*)d_in[1];           // (21,20)
    const float*         W2    = (const float*)d_in[2];           // (9,21,2)
    float*               out   = (float*)d_out;                   // (B,1)

    _Float16* btab = (_Float16*)d_ws;
    float*    w2s  = (float*)((char*)d_ws + 2048);

    const int Bsz = in_sizes[0] / N_;

    prep_kernel<<<1, 32, 0, stream>>>(Wleaf, W2, btab, w2s);
    const int blocks = (Bsz + 255) / 256;
    pc_kernel<<<blocks, 256, 0, stream>>>(x, btab, w2s, out, Bsz);
}
